// DynamicNetwork_32134945309414
// MI455X (gfx1250) — compile-verified, matrix operand order checked by reading, not run on hardware
//
#include <hip/hip_runtime.h>
#include <hip/hip_bf16.h>
#include <math.h>

#define NN 8192
#define FF 64
#define HH 128

typedef __attribute__((ext_vector_type(16))) _Float16 v16h;
typedef __attribute__((ext_vector_type(8)))  _Float16 v8h;
typedef __attribute__((ext_vector_type(8)))  float    v8f;

// ---------------------------------------------------------------------------
// Packed fragment storage in LDS: layout [frag][chunk(2)][lane(32)][8 x f16].
// Each lane reads its 16 f16 fragment elements as two contiguous 16B chunks
// (2 x ds_load_b128, conflict-free: 32 lanes x 16B = 512B = all 64 banks x2).
//
// CDNA5 wave32 f16 fragment layouts (CDNA5 ISA 7.12.2):
//   A (16x32): lane = m + 16*half, element e covers kk = (e&7) + 8*half + 16*(e>>3)
//   B (32x16): lane = n + 16*half, same kk mapping.
// Inverse (used at fill time): for kk in [0,32):
//   half = (kk>>3)&1, chunk = kk>>4, elem = kk&7.
// ---------------------------------------------------------------------------
__device__ __forceinline__ v16h load_frag(const _Float16* t, int f, int lane) {
    const v8h* p0 = (const v8h*)(t + (((size_t)(f * 2 + 0) * 32 + lane) << 3));
    const v8h* p1 = (const v8h*)(t + (((size_t)(f * 2 + 1) * 32 + lane) << 3));
    v8h lo = *p0;
    v8h hi = *p1;
    return __builtin_shufflevector(lo, hi, 0, 1, 2, 3, 4, 5, 6, 7,
                                   8, 9, 10, 11, 12, 13, 14, 15);
}

// destination index within a packed fragment region
__device__ __forceinline__ int frag_idx(int f, int kk, int lane_lo) {
    int half  = (kk >> 3) & 1;
    int chunk = kk >> 4;
    int lane  = lane_lo + (half << 4);
    return ((f * 2 + chunk) * 32 + lane) * 8 + (kk & 7);
}

// ---------------- Kernel 0: zero accumulators ----------------
__global__ void zero_kernel(float* p, int n) {
    int i = blockIdx.x * blockDim.x + threadIdx.x;
    if (i < n) p[i] = 0.0f;
}

// ---------------- Kernel 1: z_on + msg via WMMA ----------------
#define WAVES 2

__global__ __launch_bounds__(WAVES * 32)
void rows_kernel(const float* __restrict__ geom, const float* __restrict__ W_on,
                 const float* __restrict__ B_on, const float* __restrict__ W_int,
                 const float* __restrict__ B_int, float* __restrict__ msg,
                 float* __restrict__ corr) {
    // B-operand fragments: W_on has 2 k-blocks x 8 n-tiles, W_int 4 x 8.
    __shared__ __align__(16) _Float16 sWonP[16 * 512];    // 16 KB
    __shared__ __align__(16) _Float16 sWintP[32 * 512];   // 32 KB
    __shared__ float sBon[HH];
    __shared__ float sBint[HH];
    // A-operand fragments (per wave): geom tile 2 k-blocks, z_on tile 4 k-blocks.
    __shared__ __align__(16) _Float16 sAP[WAVES][2 * 512];   // 4 KB
    __shared__ __align__(16) _Float16 sZP[WAVES][4 * 512];   // 8 KB

    const int tid  = threadIdx.x;
    const int nthr = WAVES * 32;

    // Fill W_on fragments: element (k,n) -> frag f = (k>>5)*8 + (n>>4)
    for (int i = tid; i < FF * HH; i += nthr) {
        int k = i >> 7, n = i & (HH - 1);
        sWonP[frag_idx((k >> 5) * 8 + (n >> 4), k & 31, n & 15)] = (_Float16)W_on[i];
    }
    for (int i = tid; i < HH * HH; i += nthr) {
        int k = i >> 7, n = i & (HH - 1);
        sWintP[frag_idx((k >> 5) * 8 + (n >> 4), k & 31, n & 15)] = (_Float16)W_int[i];
    }
    for (int i = tid; i < HH; i += nthr) { sBon[i] = B_on[i]; sBint[i] = B_int[i]; }

    const int wave = tid >> 5;
    const int lane = tid & 31;
    const int m0   = (blockIdx.x * WAVES + wave) * 16;

    // Fill geom A-fragments: element (m,k) -> frag f = k>>5, lane_lo = m
    for (int i = lane; i < 16 * FF; i += 32) {
        int m = i >> 6, k = i & (FF - 1);
        sAP[wave][frag_idx(k >> 5, k & 31, m)] = (_Float16)geom[(size_t)(m0 + m) * FF + k];
    }
    __syncthreads();

    const int nloc  = lane & 15;
    const int mbase = (lane >> 4) << 3;   // D-layout: rows mbase..mbase+7, col nloc

    // ---- GEMM1: z_pre = geom_tile @ W_on (16x64 @ 64x128), bias + softplus ----
    const v16h ga0 = load_frag(sAP[wave], 0, lane);
    const v16h ga1 = load_frag(sAP[wave], 1, lane);
    for (int nt = 0; nt < 8; ++nt) {
        v8f acc = {};
        acc = __builtin_amdgcn_wmma_f32_16x16x32_f16(false, ga0, false,
                load_frag(sWonP, 0 * 8 + nt, lane), (short)0, acc, false, false);
        acc = __builtin_amdgcn_wmma_f32_16x16x32_f16(false, ga1, false,
                load_frag(sWonP, 1 * 8 + nt, lane), (short)0, acc, false, false);
        const int n = nt * 16 + nloc;
        const float bias = sBon[n];
        float colsum = 0.0f;
#pragma unroll
        for (int r = 0; r < 8; ++r) {
            float x  = acc[r] + bias;
            // stable softplus: max(x,0) + log(1 + exp(-|x|)) (v_exp_f32 + v_log_f32)
            float sp = fmaxf(x, 0.0f) + __logf(1.0f + __expf(-fabsf(x)));
            colsum  += sp;
            // scatter into A-fragment order for GEMM2: row m, k = n
            sZP[wave][frag_idx(n >> 5, n & 31, mbase + r)] = (_Float16)sp;
        }
        colsum += __shfl_xor(colsum, 16, 32);       // fold the two 8-row halves
        if (lane < 16) atomicAdd(&corr[n], colsum); // sum_i z_on[i,n]
    }
    __syncthreads();

    // ---- GEMM2: msg = z_on_tile @ W_int (16x128 @ 128x128) + B_int ----
    const v16h za0 = load_frag(sZP[wave], 0, lane);
    const v16h za1 = load_frag(sZP[wave], 1, lane);
    const v16h za2 = load_frag(sZP[wave], 2, lane);
    const v16h za3 = load_frag(sZP[wave], 3, lane);
    for (int nt = 0; nt < 8; ++nt) {
        v8f acc = {};
        acc = __builtin_amdgcn_wmma_f32_16x16x32_f16(false, za0, false,
                load_frag(sWintP, 0 * 8 + nt, lane), (short)0, acc, false, false);
        acc = __builtin_amdgcn_wmma_f32_16x16x32_f16(false, za1, false,
                load_frag(sWintP, 1 * 8 + nt, lane), (short)0, acc, false, false);
        acc = __builtin_amdgcn_wmma_f32_16x16x32_f16(false, za2, false,
                load_frag(sWintP, 2 * 8 + nt, lane), (short)0, acc, false, false);
        acc = __builtin_amdgcn_wmma_f32_16x16x32_f16(false, za3, false,
                load_frag(sWintP, 3 * 8 + nt, lane), (short)0, acc, false, false);
        const int n = nt * 16 + nloc;
        const float bias = sBint[n];
#pragma unroll
        for (int r = 0; r < 8; ++r)
            msg[(size_t)(m0 + mbase + r) * HH + n] = acc[r] + bias;
    }
}

// ---------------- Kernel 2: s[j] = sum_i sens(dist[i,j]) ----------------
// 268 MB streaming pass -> roofline-dominant (~11.5us at 23.3 TB/s).
#define ROWBLK 64
__global__ __launch_bounds__(256)
void sens_kernel(const float4* __restrict__ dist4, float* __restrict__ s) {
    const int j4 = blockIdx.x * 256 + threadIdx.x;   // float4 column index
    const int i0 = blockIdx.y * (NN / ROWBLK);
    float a0 = 0.f, a1 = 0.f, a2 = 0.f, a3 = 0.f;
    for (int i = i0; i < i0 + NN / ROWBLK; ++i) {
        float4 d = dist4[(size_t)i * (NN / 4) + j4];
        { float t = __builtin_amdgcn_rcpf(d.x) - 1.0f; float v = __expf(-2.0f * t * t);
          a0 += (d.x < 0.5f) ? v : 0.0f; }
        { float t = __builtin_amdgcn_rcpf(d.y) - 1.0f; float v = __expf(-2.0f * t * t);
          a1 += (d.y < 0.5f) ? v : 0.0f; }
        { float t = __builtin_amdgcn_rcpf(d.z) - 1.0f; float v = __expf(-2.0f * t * t);
          a2 += (d.z < 0.5f) ? v : 0.0f; }
        { float t = __builtin_amdgcn_rcpf(d.w) - 1.0f; float v = __expf(-2.0f * t * t);
          a3 += (d.w < 0.5f) ? v : 0.0f; }
    }
    float* sp = s + (size_t)j4 * 4;
    atomicAdd(sp + 0, a0);
    atomicAdd(sp + 1, a1);
    atomicAdd(sp + 2, a2);
    atomicAdd(sp + 3, a3);
}

// ---------------- Kernel 3: corr[h] += sum_j s[j]*msg[j,h] ----------------
__global__ __launch_bounds__(128)
void dot_kernel(const float* __restrict__ s, const float* __restrict__ msg,
                float* __restrict__ corr) {
    __shared__ float ss[128];
    const int h  = threadIdx.x;
    const int j0 = blockIdx.x * 128;
    ss[h] = s[j0 + h];
    __syncthreads();
    float acc = 0.0f;
#pragma unroll 4
    for (int j = 0; j < 128; ++j)
        acc += ss[j] * msg[(size_t)(j0 + j) * HH + h];
    atomicAdd(&corr[h], acc);
}

// ---------------- Kernel 4: outputs ----------------
__global__ __launch_bounds__(128)
void finalize_kernel(const float* __restrict__ ppp, const float* __restrict__ corr,
                     float* __restrict__ out) {
    __shared__ float red[128];
    const int h = threadIdx.x;
    float c = corr[h];
    out[h]  = ppp[h] + c;
    red[h]  = c * c;
    __syncthreads();
    for (int off = 64; off > 0; off >>= 1) {
        if (h < off) red[h] += red[h + off];
        __syncthreads();
    }
    if (h == 0) out[HH] = 0.01f * sqrtf(red[0]);
}

extern "C" void kernel_launch(void* const* d_in, const int* in_sizes, int n_in,
                              void* d_out, int out_size, void* d_ws, size_t ws_size,
                              hipStream_t stream) {
    const float* geom  = (const float*)d_in[0];
    const float* dist  = (const float*)d_in[1];
    const float* W_on  = (const float*)d_in[2];
    const float* B_on  = (const float*)d_in[3];
    const float* W_int = (const float*)d_in[4];
    const float* B_int = (const float*)d_in[5];
    const float* ppp   = (const float*)d_in[6];
    float* out = (float*)d_out;

    char* ws = (char*)d_ws;
    float* msg  = (float*)ws;                                     // N*H f32 (4 MB)
    float* s    = (float*)(ws + (size_t)NN * HH * sizeof(float)); // N f32
    float* corr = s + NN;                                         // H f32 (contiguous)

    zero_kernel<<<(NN + HH + 255) / 256, 256, 0, stream>>>(s, NN + HH);
    rows_kernel<<<NN / (WAVES * 16), WAVES * 32, 0, stream>>>(geom, W_on, B_on,
                                                              W_int, B_int, msg, corr);
    sens_kernel<<<dim3(NN / 1024, ROWBLK), 256, 0, stream>>>((const float4*)dist, s);
    dot_kernel<<<NN / 128, 128, 0, stream>>>(s, msg, corr);
    finalize_kernel<<<1, 128, 0, stream>>>(ppp, corr, out);
}
